// RecurTreeGen_67070209294950
// MI455X (gfx1250) — compile-verified, matrix operand order checked by reading, not run on hardware
//
#include <hip/hip_runtime.h>
#include <hip/hip_bf16.h>

typedef __attribute__((ext_vector_type(16))) __bf16 v16bf;
typedef __attribute__((ext_vector_type(8)))  __bf16 v8bf;
typedef __attribute__((ext_vector_type(8)))  float  v8f;
typedef __attribute__((ext_vector_type(4)))  unsigned int v4u;
typedef __attribute__((ext_vector_type(8)))  int v8i;
typedef __attribute__((ext_vector_type(4)))  int v4i;

#define D 64
#define FIVE_D 320
#define N_TILES 20          // 320 / 16
#define K_TILES 4           // 128 / 32
#define PACK_ELEMS (N_TILES * K_TILES * 32 * 16)   // 40960 bf16 = 81920 B
#define TAIL_N 256          // levels with n <= TAIL_N run LDS-resident
#define TOTAL_LEVELS 19

__device__ __forceinline__ float sigf(float x) {
    return 1.0f / (1.0f + __expf(-x));
}

// ---------------------------------------------------------------------------
// Tensor Data Mover staging: 1-row tile of n4 dwords, global -> LDS.
// D# built per CDNA5 ISA (group0: count/lds_addr/global_addr/type,
// group1: data_size=4B, tile_dim0=tensor_dim0=stride=n4, dims1=1).
// ---------------------------------------------------------------------------
#if __has_builtin(__builtin_amdgcn_tensor_load_to_lds) && \
    __has_builtin(__builtin_amdgcn_s_wait_tensorcnt)
#define HAS_TDM 1
__device__ __forceinline__ void tdm_load_1d(unsigned int lds_off,
                                            const void* gsrc,
                                            unsigned int n4) {
    unsigned long long ga = (unsigned long long)(size_t)gsrc;
    v4u g0;
    g0[0] = 1u;                                           // count=1, no gather
    g0[1] = lds_off;                                      // lds_addr (bytes)
    g0[2] = (unsigned int)ga;                             // global_addr[31:0]
    g0[3] = (unsigned int)((ga >> 32) & 0x01FFFFFFu)      // global_addr[56:32]
            | 0x80000000u;                                // type=2 ("image")
    v8i g1;
    g1[0] = (int)(2u << 16);                              // data_size = 4 B
    g1[1] = (int)((n4 & 0xFFFFu) << 16);                  // tensor_dim0 lo
    g1[2] = (int)(((n4 >> 16) & 0xFFFFu) | (1u << 16));   // dim0 hi | tensor_dim1=1
    g1[3] = (int)((n4 & 0xFFFFu) << 16);                  // tile_dim0 = n4
    g1[4] = 1;                                            // tile_dim1 = 1
    g1[5] = (int)n4;                                      // tensor_dim0_stride lo
    g1[6] = (int)((n4 >> 16) & 0xFFFFu);                  // stride hi
    g1[7] = 0;
    v4i z = {0, 0, 0, 0};
#if __clang_major__ >= 23
    v8i z8 = {0, 0, 0, 0, 0, 0, 0, 0};
    __builtin_amdgcn_tensor_load_to_lds(g0, g1, z, z, z8, 0);
#else
    __builtin_amdgcn_tensor_load_to_lds(g0, g1, z, z, 0);
#endif
}
#else
#define HAS_TDM 0
#endif

// ---------------------------------------------------------------------------
// Pack W = [Wl; Wr]  (128 x 320, row major fp32) into the exact per-lane
// v16bf B-fragment layout of V_WMMA_F32_16X16X32_BF16:
//   fragment index f = (ntile*4 + ktile)*32 + lane, 16 bf16 elems per lane.
//   elem j -> K = ktile*32 + (j<8 ? kgrp*8 + j : 16 + kgrp*8 + (j-8)),
//   col N = ntile*16 + (lane&15), kgrp = lane>>4.
// ---------------------------------------------------------------------------
__global__ void pack_weights(const float* __restrict__ Wl,
                             const float* __restrict__ Wr,
                             __bf16* __restrict__ pack) {
    int id = blockIdx.x * blockDim.x + threadIdx.x;
    if (id >= N_TILES * K_TILES * 32) return;
    int lane  = id & 31;
    int kt    = (id >> 5) & 3;
    int ntile = id >> 7;
    int kgrp  = lane >> 4;
    int n     = ntile * 16 + (lane & 15);
#pragma unroll
    for (int j = 0; j < 16; ++j) {
        int k = kt * 32 + (j < 8 ? kgrp * 8 + j : 16 + kgrp * 8 + (j - 8));
        float v = (k < D) ? Wl[k * FIVE_D + n] : Wr[(k - D) * FIVE_D + n];
        pack[(size_t)id * 16 + j] = (__bf16)v;
    }
}

// Save last row of the current level's input (Fenwick prefix representative).
template <typename HT>
__global__ void save_last(const HT* __restrict__ h_in,
                          const float* __restrict__ c_in,
                          int n_in, int lvl,
                          __bf16* __restrict__ lastH,
                          float* __restrict__ lastC) {
    int t = threadIdx.x;  // 64 threads
    size_t base = (size_t)(n_in - 1) * D + t;
    lastH[lvl * D + t] = (__bf16)(float)h_in[base];
    lastC[lvl * D + t] = c_in[base];
}

// ---------------------------------------------------------------------------
// One 16-row output tile of a Fenwick merge:
//   A row m (128 wide) = [h[2m] | h[2m+1]] -> contiguous in memory/LDS.
//   g = A @ Wpack (+bias via WMMA C operand), then LSTM gates.
// Pointers are generic: works on both global buffers and LDS-resident state.
// ---------------------------------------------------------------------------
template <typename HT>
__device__ __forceinline__ void tree_tile(
    const HT* h_src, const float* c_src,
    const __bf16* wfrag, const float* bias,
    __bf16* h_dst, float* c_dst,
    int n_out, int m0, int lane) {

    const int mrow = lane & 15;
    const int kgrp = lane >> 4;

    // ---- A fragments (16 rows x 128 K, bf16) -------------------------------
    v16bf a[K_TILES];
    int ar = m0 + mrow;
    if (ar > n_out - 1) ar = n_out - 1;              // clamp partial tile
    const HT* arow = h_src + (size_t)(2 * ar) * D;   // 128 contiguous elems
#pragma unroll
    for (int kt = 0; kt < K_TILES; ++kt) {
        int k0 = kt * 32 + kgrp * 8;
        int k1 = kt * 32 + 16 + kgrp * 8;
        if constexpr (sizeof(HT) == 4) {             // fp32 input (level 0)
            const float4* p0 = reinterpret_cast<const float4*>(arow + k0);
            const float4* p1 = reinterpret_cast<const float4*>(arow + k1);
            float4 f0 = p0[0], f1 = p0[1], f2 = p1[0], f3 = p1[1];
            a[kt][0]  = (__bf16)f0.x; a[kt][1]  = (__bf16)f0.y;
            a[kt][2]  = (__bf16)f0.z; a[kt][3]  = (__bf16)f0.w;
            a[kt][4]  = (__bf16)f1.x; a[kt][5]  = (__bf16)f1.y;
            a[kt][6]  = (__bf16)f1.z; a[kt][7]  = (__bf16)f1.w;
            a[kt][8]  = (__bf16)f2.x; a[kt][9]  = (__bf16)f2.y;
            a[kt][10] = (__bf16)f2.z; a[kt][11] = (__bf16)f2.w;
            a[kt][12] = (__bf16)f3.x; a[kt][13] = (__bf16)f3.y;
            a[kt][14] = (__bf16)f3.z; a[kt][15] = (__bf16)f3.w;
        } else {                                     // bf16 input
            v8bf lo = *reinterpret_cast<const v8bf*>(arow + k0);
            v8bf hi = *reinterpret_cast<const v8bf*>(arow + k1);
#pragma unroll
            for (int j = 0; j < 8; ++j) { a[kt][j] = lo[j]; a[kt][8 + j] = hi[j]; }
        }
    }

    const v16bf* pw = reinterpret_cast<const v16bf*>(wfrag);
    const int colg_base = lane & 15;

    // ---- per 16-column slice t: gate g lives in N-tile (g*4 + t) -----------
#pragma unroll
    for (int t = 0; t < 4; ++t) {
        int colg = t * 16 + colg_base;
        v8f acc[5];
#pragma unroll
        for (int g = 0; g < 5; ++g) {
            float bv = bias[g * D + colg];
            v8f c;
#pragma unroll
            for (int r = 0; r < 8; ++r) c[r] = bv;
            int ntile = g * 4 + t;
#pragma unroll
            for (int kt = 0; kt < K_TILES; ++kt) {
                v16bf bf = pw[(size_t)(ntile * K_TILES + kt) * 32 + lane];
                c = __builtin_amdgcn_wmma_f32_16x16x32_bf16(
                        false, a[kt], false, bf, (short)0, c, false, false);
            }
            acc[g] = c;
        }
        // D layout: (vgpr r, lane) -> row r + 8*kgrp, col lane&15
#pragma unroll
        for (int r = 0; r < 8; ++r) {
            int row = m0 + r + kgrp * 8;
            bool act = row < n_out;
            int rowc = act ? row : n_out - 1;
            size_t il = (size_t)(2 * rowc) * D + colg;
            float cl = c_src[il];
            float cr = c_src[il + D];
            float iv = acc[0][r], ov = acc[1][r], uv = acc[2][r];
            float flv = acc[3][r], frv = acc[4][r];
            float cn = sigf(iv) * tanhf(uv) + sigf(flv) * cl + sigf(frv) * cr;
            float hn = sigf(ov) * tanhf(cn);
            if (act) {
                size_t o = (size_t)row * D + colg;
                h_dst[o] = (__bf16)hn;
                c_dst[o] = cn;
            }
        }
    }
}

// ---------------------------------------------------------------------------
// Bulk merge level (n_out >= TAIL_N): packed weights staged to LDS once per
// block via the Tensor Data Mover (8 waves share -> 8x less L2 traffic).
// ---------------------------------------------------------------------------
template <typename HT>
__global__ __launch_bounds__(256) void merge_level(
    const HT* __restrict__ h_in, const float* __restrict__ c_in,
    const __bf16* __restrict__ packW, const float* __restrict__ bias,
    __bf16* __restrict__ h_out, float* __restrict__ c_out, int n_out) {

    __shared__ __bf16 w_lds[PACK_ELEMS];
    const int tid = threadIdx.x;
#if HAS_TDM
    if (tid < 32) {                           // wave 0 issues the DMA
        tdm_load_1d((unsigned int)(size_t)w_lds, packW, PACK_ELEMS / 2);
        __builtin_amdgcn_s_wait_tensorcnt(0);
    }
#else
    for (int i = tid; i < PACK_ELEMS / 8; i += 256)
        ((uint4*)w_lds)[i] = ((const uint4*)packW)[i];
#endif
    __syncthreads();

    int wave = (int)((blockIdx.x * blockDim.x + tid) >> 5);
    int lane = tid & 31;
    int m0 = wave * 16;
    if (m0 >= n_out) return;  // wave-uniform

    tree_tile<HT>(h_in, c_in, w_lds, bias, h_out, c_out, n_out, m0, lane);
}

// ---------------------------------------------------------------------------
// Fused tail: all levels with n <= TAIL_N plus the 19-step summary fold,
// single block, state fully LDS-resident (ping-pong regions, no HBM traffic).
// ---------------------------------------------------------------------------
__global__ __launch_bounds__(256) void tail_kernel(
    const __bf16* __restrict__ h_in, const float* __restrict__ c_in,
    const __bf16* __restrict__ packW, const float* __restrict__ biasM,
    const __bf16* __restrict__ lastH_g, const float* __restrict__ lastC_g,
    int nsaved,
    const float* __restrict__ sWl, const float* __restrict__ sWr,
    const float* __restrict__ sB, float* __restrict__ out) {

    __shared__ __bf16 h_lds[(TAIL_N + TAIL_N / 2) * D];   //  48 KB ping-pong
    __shared__ float  c_lds[(TAIL_N + TAIL_N / 2) * D];   //  96 KB ping-pong
    __shared__ __bf16 w_lds[PACK_ELEMS];                  //  80 KB
    __shared__ __bf16 lh_lds[TOTAL_LEVELS * D];
    __shared__ float  lc_lds[TOTAL_LEVELS * D];
    __shared__ float  g_s[FIVE_D];
    __shared__ float  sh_s[D], sc_s[D];

    const int tid = threadIdx.x;

    // ---- stage level state + weights + saved snapshots into LDS ------------
#if HAS_TDM
    if (tid < 32) {                           // wave 0 issues all DMAs
        tdm_load_1d((unsigned int)(size_t)h_lds, h_in, TAIL_N * D / 2);
        tdm_load_1d((unsigned int)(size_t)c_lds, c_in, TAIL_N * D);
        tdm_load_1d((unsigned int)(size_t)w_lds, packW, PACK_ELEMS / 2);
        tdm_load_1d((unsigned int)(size_t)lh_lds, lastH_g,
                    (unsigned int)(nsaved * D / 2));
        tdm_load_1d((unsigned int)(size_t)lc_lds, lastC_g,
                    (unsigned int)(nsaved * D));
        __builtin_amdgcn_s_wait_tensorcnt(0);
    }
#else
    for (int i = tid; i < TAIL_N * D / 8; i += 256)       // h: bf16 x8 = 16 B
        ((uint4*)h_lds)[i] = ((const uint4*)h_in)[i];
    for (int i = tid; i < TAIL_N * D / 4; i += 256)       // c: f32 x4 = 16 B
        ((float4*)c_lds)[i] = ((const float4*)c_in)[i];
    for (int i = tid; i < PACK_ELEMS / 8; i += 256)
        ((uint4*)w_lds)[i] = ((const uint4*)packW)[i];
    for (int i = tid; i < nsaved * D; i += 256) {
        lh_lds[i] = lastH_g[i];
        lc_lds[i] = lastC_g[i];
    }
#endif
    __syncthreads();

    const int wave = tid >> 5;
    const int lane = tid & 31;
    const int m0 = wave * 16;

    // ---- LDS-resident merge levels ----------------------------------------
    int n = TAIL_N;
    int in_off = 0;
    int lvl = nsaved;
    while (n > 1) {
        int n_out = n >> 1;
        int out_off = (in_off == 0) ? TAIL_N * D : 0;
        if (tid < D) {  // waves 0-1 whole: wave-uniform branch
            lh_lds[lvl * D + tid] = h_lds[in_off + (n - 1) * D + tid];
            lc_lds[lvl * D + tid] = c_lds[in_off + (n - 1) * D + tid];
        }
        if (m0 < n_out) {
            tree_tile<__bf16>(h_lds + in_off, c_lds + in_off, w_lds, biasM,
                              h_lds + out_off, c_lds + out_off,
                              n_out, m0, lane);
        }
        __syncthreads();
        n = n_out;
        in_off = out_off;
        ++lvl;
    }

    // ---- summary fold over TOTAL_LEVELS representatives --------------------
    if (tid < D) {
        sh_s[tid] = (float)h_lds[in_off + tid];  // root h
        sc_s[tid] = c_lds[in_off + tid];         // root c
    }
    __syncthreads();
    for (int l = TOTAL_LEVELS - 1; l >= 0; --l) {
        for (int nn = tid; nn < FIVE_D; nn += 256) {
            float acc = sB[nn];
#pragma unroll 8
            for (int k = 0; k < D; ++k)
                acc += sh_s[k] * sWl[k * FIVE_D + nn] +
                       (float)lh_lds[l * D + k] * sWr[k * FIVE_D + nn];
            g_s[nn] = acc;
        }
        __syncthreads();
        if (tid < D) {
            float iv = g_s[tid], ov = g_s[D + tid], uv = g_s[2 * D + tid];
            float flv = g_s[3 * D + tid], frv = g_s[4 * D + tid];
            float cn = sigf(iv) * tanhf(uv) + sigf(flv) * sc_s[tid] +
                       sigf(frv) * lc_lds[l * D + tid];
            sc_s[tid] = cn;
            sh_s[tid] = sigf(ov) * tanhf(cn);
        }
        __syncthreads();
    }
    if (tid < D) { out[tid] = sh_s[tid]; out[D + tid] = sc_s[tid]; }
}

// ---------------------------------------------------------------------------
extern "C" void kernel_launch(void* const* d_in, const int* in_sizes, int n_in_args,
                              void* d_out, int out_size, void* d_ws, size_t ws_size,
                              hipStream_t stream) {
    const float* h_bot = (const float*)d_in[0];
    const float* c_bot = (const float*)d_in[1];
    const float* mWl   = (const float*)d_in[2];
    const float* mWr   = (const float*)d_in[3];
    const float* mB    = (const float*)d_in[4];
    const float* sWl   = (const float*)d_in[5];
    const float* sWr   = (const float*)d_in[6];
    const float* sB    = (const float*)d_in[7];
    float* out = (float*)d_out;

    const int N0 = in_sizes[0] / D;   // 524288 leaves

    // ---- workspace layout --------------------------------------------------
    char* ws = (char*)d_ws;
    __bf16* packM = (__bf16*)(ws);                    //  81920 B
    __bf16* lastH = (__bf16*)(ws + 81920);            //  <=4096 B
    float*  lastC = (float*) (ws + 86016);            //  <=8192 B
    char* bufA = ws + 94208;                          // level-out ping
    const size_t HA_BYTES = (size_t)(N0 / 2) * D * sizeof(__bf16);   // 32 MB
    const size_t CA_BYTES = (size_t)(N0 / 2) * D * sizeof(float);    // 64 MB
    char* bufB = bufA + HA_BYTES + CA_BYTES;          // level-out pong (half)
    __bf16* hA = (__bf16*)bufA;  float* cA = (float*)(bufA + HA_BYTES);
    __bf16* hB = (__bf16*)bufB;  float* cB = (float*)(bufB + HA_BYTES / 2);

    // ---- pre-pack [Wl;Wr] into WMMA B-fragment layout ----------------------
    pack_weights<<<10, 256, 0, stream>>>(mWl, mWr, packM);

    // ---- level 0 (fp32 inputs) ---------------------------------------------
    int n = N0, lvl = 0;
    save_last<float><<<1, D, 0, stream>>>(h_bot, c_bot, n, lvl, lastH, lastC);
    int n_out = n >> 1;
    {
        int tiles = (n_out + 15) / 16;
        int blocks = (tiles + 7) / 8;
        merge_level<float><<<blocks, 256, 0, stream>>>(
            h_bot, c_bot, packM, mB, hA, cA, n_out);
    }
    const __bf16* hin = hA;
    const float*  cin = cA;
    n = n_out;
    lvl = 1;
    bool useB = true;

    // ---- bulk levels until n == TAIL_N (bf16 h, fp32 c) --------------------
    while (n > TAIL_N) {
        save_last<__bf16><<<1, D, 0, stream>>>(hin, cin, n, lvl, lastH, lastC);
        n_out = n >> 1;
        __bf16* ho = useB ? hB : hA;
        float*  co = useB ? cB : cA;
        int tiles = (n_out + 15) / 16;
        int blocks = (tiles + 7) / 8;
        merge_level<__bf16><<<blocks, 256, 0, stream>>>(
            hin, cin, packM, mB, ho, co, n_out);
        hin = ho; cin = co; n = n_out; ++lvl; useB = !useB;
    }

    // ---- fused LDS-resident tail + summary ---------------------------------
    tail_kernel<<<1, 256, 0, stream>>>(
        hin, cin, packM, mB, lastH, lastC, lvl, sWl, sWr, sB, out);
}
